// Supervised_2319282340435
// MI455X (gfx1250) — compile-verified, matrix operand order checked by reading, not run on hardware
//
#include <hip/hip_runtime.h>

#define EPSF 1e-5f

typedef __bf16 bf16_t;
typedef bf16_t v16bf __attribute__((ext_vector_type(16)));
typedef float  v8f   __attribute__((ext_vector_type(8)));
typedef unsigned int v4u __attribute__((ext_vector_type(4)));

union FragBF {
  v16bf  bf;
  v4u    u[2];
  bf16_t e[16];
};

__device__ __forceinline__ unsigned short f2bfu(float f) {
  unsigned u = __float_as_uint(f);
  u += 0x7FFFu + ((u >> 16) & 1u);            // round-to-nearest-even
  return (unsigned short)(u >> 16);
}
__device__ __forceinline__ float bfu2f(unsigned short u) {
  return __uint_as_float(((unsigned)u) << 16);
}
__device__ __forceinline__ float sigf(float x) { return 1.f / (1.f + __expf(-x)); }

// ---- CDNA5 async global->LDS helpers (ASYNCcnt path, ISA 15.18.3 op 98) ----
__device__ __forceinline__ unsigned lds_off_u32(const void* p) {
  // generic -> LDS(AS3) pointer -> 32-bit wave-relative LDS byte address
  return (unsigned)(unsigned long long)(const __attribute__((address_space(3))) void*)p;
}
__device__ __forceinline__ void async_ld16(unsigned dst_lds, const unsigned short* src) {
  asm volatile("global_load_async_to_lds_b128 %0, %1, off"
               :: "v"(dst_lds), "v"(src)
               : "memory");
}
__device__ __forceinline__ void wait_async0() {
  asm volatile("s_wait_asynccnt 0" ::: "memory");
}

// Fragment loader for 16x32 bf16 A (and, by lane<->N symmetry, 32x16 B from
// W[N][K] row-major).  ISA 7.12.2: lanes 0-15 hold K[k0..k0+7] / K[k0+16..k0+23],
// lanes 16-31 hold K[k0+8..k0+15] / K[k0+24..k0+31].
__device__ __forceinline__ v16bf load_frag(const unsigned short* __restrict__ base,
                                           int ld, int lane, int k0) {
  int r  = lane & 15;
  int kb = k0 + ((lane >> 4) << 3);
  const unsigned short* p = base + (size_t)r * ld + kb;
  FragBF f;
  f.u[0] = *(const v4u*)(p);
  f.u[1] = *(const v4u*)(p + 16);
  return f.bf;
}

// B fragment from an LDS-staged [rows][32] bf16 tile (row-major, 64B rows).
__device__ __forceinline__ v16bf frag_from_lds_bf(const unsigned short* base,
                                                  int row0, int lane) {
  int r  = lane & 15;
  int kb = (lane >> 4) << 3;                   // 0 or 8 halfwords
  const unsigned short* p = base + (row0 + r) * 32 + kb;
  FragBF f;
  f.u[0] = *(const v4u*)(p);                   // K kb..kb+7
  f.u[1] = *(const v4u*)(p + 16);              // K kb+16..kb+23
  return f.bf;
}

// A fragment built on the fly from an LDS-resident f32 matrix [16][256].
__device__ __forceinline__ v16bf frag_from_lds_f32(const float* __restrict__ src,
                                                   int lane, int k0) {
  int r  = lane & 15;
  int kb = k0 + ((lane >> 4) << 3);
  FragBF f;
#pragma unroll
  for (int i = 0; i < 8; ++i) f.e[i]     = (bf16_t)src[r * 256 + kb + i];
#pragma unroll
  for (int i = 0; i < 8; ++i) f.e[8 + i] = (bf16_t)src[r * 256 + kb + 16 + i];
  return f.bf;
}

#define WMMA_BF16(a, b, c) \
  __builtin_amdgcn_wmma_f32_16x16x32_bf16(false, (a), false, (b), (short)0, (c), false, false)

// ---------------- shapes ----------------
#define BB 16
#define TT 60
#define NN 40
#define CC 1024
#define HH 256
#define RR 38400   // B*T*N

// ---------------- ws layout (bytes) ----------------
#define OFF_X    ((size_t)0)                       // 38400*1024 bf16   (aliased by IP f32 later)
#define OFF_H1   ((size_t)78643200)                // 38400*1024 bf16
#define OFF_OBJ  ((size_t)157286400)               // 38400*256  bf16
#define OFF_MASK ((size_t)176947200)               // 38400 f32
#define OFF_W1   ((size_t)177100800)               // 1024*1024 bf16
#define OFF_W2   ((size_t)179197952)               // 256*1024 bf16
#define OFF_WIH  ((size_t)179722240)               // 1024*256 bf16
#define OFF_WHH  ((size_t)180246528)               // 1024*256 bf16
#define OFF_UAT  ((size_t)180770816)               // 256*256 bf16 (transposed)
#define OFF_WAT  ((size_t)180901888)               // 256*256 bf16 (transposed)
#define OFF_SC1  ((size_t)181032960)               // 1024 f32
#define OFF_SH1  ((size_t)181037056)               // 1024 f32

// ---------------------------------------------------------------------------
// Kernel 1: mask (sum over raw C != 0) + BN1 + f32->bf16.  One block per row.
// ---------------------------------------------------------------------------
__global__ __launch_bounds__(256) void k_prep_x(
    const float* __restrict__ xf, const float* __restrict__ g,
    const float* __restrict__ be, const float* __restrict__ mu,
    const float* __restrict__ var, unsigned short* __restrict__ xb,
    float* __restrict__ mask) {
  __shared__ float red[256];
  int r = blockIdx.x, tid = threadIdx.x, c = tid * 4;
  const float* row = xf + (size_t)r * CC;
  float4 x  = *(const float4*)(row + c);
  float4 gg = *(const float4*)(g + c);
  float4 bb = *(const float4*)(be + c);
  float4 mm = *(const float4*)(mu + c);
  float4 vv = *(const float4*)(var + c);
  float s = x.x + x.y + x.z + x.w;
  unsigned short y0 = f2bfu((x.x - mm.x) * rsqrtf(vv.x + EPSF) * gg.x + bb.x);
  unsigned short y1 = f2bfu((x.y - mm.y) * rsqrtf(vv.y + EPSF) * gg.y + bb.y);
  unsigned short y2 = f2bfu((x.z - mm.z) * rsqrtf(vv.z + EPSF) * gg.z + bb.z);
  unsigned short y3 = f2bfu((x.w - mm.w) * rsqrtf(vv.w + EPSF) * gg.w + bb.w);
  uint2 o;
  o.x = (unsigned)y0 | ((unsigned)y1 << 16);
  o.y = (unsigned)y2 | ((unsigned)y3 << 16);
  *(uint2*)(xb + (size_t)r * CC + c) = o;
  red[tid] = s;
  __syncthreads();
  for (int st = 128; st > 0; st >>= 1) {
    if (tid < st) red[tid] += red[tid + st];
    __syncthreads();
  }
  if (tid == 0) mask[r] = (red[0] != 0.f) ? 1.f : 0.f;
}

// ---------------------------------------------------------------------------
// Kernel 2: weight prep (bf16 copies, transposes, BN2 fold).
// ---------------------------------------------------------------------------
__global__ __launch_bounds__(256) void k_prep_w(
    const float* __restrict__ w1, const float* __restrict__ w2,
    const float* __restrict__ wih, const float* __restrict__ whh,
    const float* __restrict__ ua, const float* __restrict__ wa,
    const float* __restrict__ b1, const float* __restrict__ og,
    const float* __restrict__ ob, const float* __restrict__ om,
    const float* __restrict__ ov, unsigned short* __restrict__ W1,
    unsigned short* __restrict__ W2, unsigned short* __restrict__ WIH,
    unsigned short* __restrict__ WHH, unsigned short* __restrict__ UAT,
    unsigned short* __restrict__ WAT, float* __restrict__ SC1,
    float* __restrict__ SH1) {
  int i = blockIdx.x * blockDim.x + threadIdx.x;
  if (i < 1024 * 1024) W1[i] = f2bfu(w1[i]);
  if (i < 256 * 1024) W2[i] = f2bfu(w2[i]);
  if (i < 1024 * 256) { WIH[i] = f2bfu(wih[i]); WHH[i] = f2bfu(whh[i]); }
  if (i < 256 * 256) {
    int d = i >> 8, c = i & 255;
    UAT[d * 256 + c] = f2bfu(ua[c * 256 + d]);   // store [d][c] = att_ua[c][d]
    WAT[d * 256 + c] = f2bfu(wa[c * 256 + d]);
  }
  if (i < 1024) {
    float sc = og[i] * rsqrtf(ov[i] + EPSF);
    SC1[i] = sc;
    SH1[i] = (b1[i] - om[i]) * sc + ob[i];
  }
}

// ---------------------------------------------------------------------------
// GEMM core (K = 1024): block = 8 waves, tile 128x64.  B-tile (64x32 bf16,
// 4 KB) staged in LDS by cooperative global_load_async_to_lds_b128, double
// buffered; A fragments register double-buffered.  Per k-step:
//   s_wait_asynccnt 0 ; barrier ; issue copy(k+1) ; 4x (ds-frag + WMMA).
// Computes acc[4] (4 col-tiles of 16) for this wave's 16-row tile.
// ---------------------------------------------------------------------------
__device__ __forceinline__ void gemm_core_k1024(
    const unsigned short* __restrict__ Ain, const unsigned short* __restrict__ W,
    int rowTile, int bn, int tid, int lane, v8f acc[4]) {
  __shared__ __align__(16) unsigned short Bs[2][64 * 32];
  const unsigned short* Abase = Ain + (size_t)rowTile * CC;
  int crow = tid >> 2, cch = tid & 3;               // 256 threads copy 64x32
  const unsigned short* srcB = W + (size_t)(bn + crow) * CC + cch * 8;
  unsigned dstB0 = lds_off_u32(&Bs[0][crow * 32 + cch * 8]);
  unsigned dstB1 = lds_off_u32(&Bs[1][crow * 32 + cch * 8]);
#pragma unroll
  for (int j = 0; j < 4; ++j)
#pragma unroll
    for (int v = 0; v < 8; ++v) acc[j][v] = 0.f;
  async_ld16(dstB0, srcB);                          // k-step 0
  v16bf a = load_frag(Abase, CC, lane, 0);
  for (int kk = 0; kk < 32; ++kk) {
    wait_async0();                  // my copy of buf[kk&1] complete
    __syncthreads();                // all copies landed; all prior reads done
    if (kk + 1 < 32)
      async_ld16((kk & 1) ? dstB0 : dstB1, srcB + (kk + 1) * 32);
    v16bf a_n = (kk + 1 < 32) ? load_frag(Abase, CC, lane, (kk + 1) * 32) : a;
    const unsigned short* Bcur = Bs[kk & 1];
#pragma unroll
    for (int j = 0; j < 4; ++j) {
      v16bf b = frag_from_lds_bf(Bcur, j * 16, lane);
      acc[j] = WMMA_BF16(a, b, acc[j]);
    }
    a = a_n;
  }
}

// Kernel 3: GEMM1  h1 = relu(objbn(X @ W1^T + b1))   [38400,1024]x[1024,1024]
__global__ __launch_bounds__(256) void k_gemm1(
    const unsigned short* __restrict__ Ain, const unsigned short* __restrict__ W,
    const float* __restrict__ SC, const float* __restrict__ SH,
    unsigned short* __restrict__ H1) {
  int tid = threadIdx.x, wave = tid >> 5, lane = tid & 31;
  int rowTile = blockIdx.x * 128 + wave * 16;
  int bn = blockIdx.y * 64;
  v8f acc[4];
  gemm_core_k1024(Ain, W, rowTile, bn, tid, lane, acc);
  int roff = (lane >> 4) << 3, ncol = lane & 15;
#pragma unroll
  for (int j = 0; j < 4; ++j) {
    int n = bn + j * 16 + ncol;
    float sc = SC[n], sh = SH[n];
#pragma unroll
    for (int v = 0; v < 8; ++v) {
      int row = rowTile + v + roff;
      float val = fmaxf(acc[j][v] * sc + sh, 0.f);
      H1[(size_t)row * CC + n] = f2bfu(val);
    }
  }
}

// Kernel 4: GEMM2  obj = relu(relu(h1 @ W2^T + b2) * mask)  [38400,1024]x[1024,256]
__global__ __launch_bounds__(256) void k_gemm2(
    const unsigned short* __restrict__ Ain, const unsigned short* __restrict__ W,
    const float* __restrict__ b2, const float* __restrict__ mask,
    unsigned short* __restrict__ OBJ) {
  int tid = threadIdx.x, wave = tid >> 5, lane = tid & 31;
  int rowTile = blockIdx.x * 128 + wave * 16;
  int bn = blockIdx.y * 64;
  v8f acc[4];
  gemm_core_k1024(Ain, W, rowTile, bn, tid, lane, acc);
  int roff = (lane >> 4) << 3, ncol = lane & 15;
#pragma unroll
  for (int j = 0; j < 4; ++j) {
    int n = bn + j * 16 + ncol;
    float bias = b2[n];
#pragma unroll
    for (int v = 0; v < 8; ++v) {
      int row = rowTile + v + roff;
      float val = fmaxf(acc[j][v] + bias, 0.f) * mask[row];
      OBJ[(size_t)row * HH + n] = f2bfu(fmaxf(val, 0.f));
    }
  }
}

// ---------------------------------------------------------------------------
// Kernel 5: ip = obj @ att_ua + ba   [38400,256]x[256,256] -> f32
// ---------------------------------------------------------------------------
__global__ __launch_bounds__(256) void k_gemm_ip(
    const unsigned short* __restrict__ OBJ, const unsigned short* __restrict__ UAT,
    const float* __restrict__ ba, float* __restrict__ IP) {
  int wave = threadIdx.x >> 5, lane = threadIdx.x & 31;
  int rowTile = blockIdx.x * 128 + wave * 16;
  int bn = blockIdx.y * 64;
  const unsigned short* Abase = OBJ + (size_t)rowTile * HH;
  v8f acc[4];
#pragma unroll
  for (int j = 0; j < 4; ++j)
#pragma unroll
    for (int v = 0; v < 8; ++v) acc[j][v] = 0.f;
  for (int k0 = 0; k0 < HH; k0 += 32) {
    v16bf a = load_frag(Abase, HH, lane, k0);
#pragma unroll
    for (int j = 0; j < 4; ++j) {
      v16bf b = load_frag(UAT + (size_t)(bn + j * 16) * HH, HH, lane, k0);
      acc[j] = WMMA_BF16(a, b, acc[j]);
    }
  }
  int roff = (lane >> 4) << 3, ncol = lane & 15;
#pragma unroll
  for (int j = 0; j < 4; ++j) {
    int n = bn + j * 16 + ncol;
    float bias = ba[n];
#pragma unroll
    for (int v = 0; v < 8; ++v) {
      int row = rowTile + v + roff;
      IP[(size_t)row * HH + n] = acc[j][v] + bias;
    }
  }
}

// ---------------------------------------------------------------------------
// Kernel 6: persistent attention-LSTM scan.  One workgroup, 16 waves.
// hx/cx live in LDS; WMMA for hx@att_wa and the gates GEMMs; gate column
// assignment puts i/f/g/o for each (b,h) in one lane -> pointwise LSTM update
// straight from accumulators (no gates LDS round-trip).
// ---------------------------------------------------------------------------
__global__ __launch_bounds__(512) void k_scan(
    const unsigned short* __restrict__ OBJ, const float* __restrict__ IP,
    const unsigned short* __restrict__ WAT, const unsigned short* __restrict__ WIH,
    const unsigned short* __restrict__ WHH, const float* __restrict__ attw,
    const float* __restrict__ mask, const float* __restrict__ bih,
    const float* __restrict__ bhh, const float* __restrict__ collw,
    const float* __restrict__ collb, const float* __restrict__ riskw,
    const float* __restrict__ riskb, float* __restrict__ out) {
  __shared__ float hx[16 * 256];
  __shared__ float cx[16 * 256];
  __shared__ float qatt[16 * 256];   // q in phases A/B, attention in D/E
  __shared__ float sc_s[40 * 16];
  __shared__ float sc_a[40 * 16];
  __shared__ float tmpc[32];

  int tid = threadIdx.x, wave = tid >> 5, lane = tid & 31;
  for (int i = tid; i < 16 * 256; i += 512) { hx[i] = 0.f; cx[i] = 0.f; }
  __syncthreads();

  float* out_soft = out;                    // [B,T,2]
  float* out_logc = out + 1920;             // [T,B,2]
  float* out_sigr = out + 3840;             // [B,T,N]
  float* out_scr  = out + 3840 + 38400;     // [T,B,N]

  for (int t = 0; t < TT; ++t) {
    // ---- Phase A: q = hx @ att_wa   (wave -> one 16-col tile) ----
    {
      v8f acc;
#pragma unroll
      for (int v = 0; v < 8; ++v) acc[v] = 0.f;
      for (int k0 = 0; k0 < HH; k0 += 32) {
        v16bf a = frag_from_lds_f32(hx, lane, k0);
        v16bf b = load_frag(WAT + (size_t)wave * 16 * HH, HH, lane, k0);
        acc = WMMA_BF16(a, b, acc);
      }
      int roff = (lane >> 4) << 3, ncol = lane & 15;
#pragma unroll
      for (int v = 0; v < 8; ++v)
        qatt[(v + roff) * 256 + wave * 16 + ncol] = acc[v];
    }
    __syncthreads();
    // ---- Phase B: scores[n][b] = sum_d tanh(q[b,d]+ip[b,t,n,d]) * attw[d] ----
    for (int p = wave; p < 640; p += 16) {
      int n = p >> 4, b = p & 15;
      const float* ipr = IP + (size_t)(((b * TT) + t) * NN + n) * HH;
      const float* qr = qatt + b * 256;
      float s = 0.f;
      for (int d = lane; d < HH; d += 32) s += tanhf(qr[d] + ipr[d]) * attw[d];
      for (int off = 16; off; off >>= 1) s += __shfl_down(s, off, 32);
      if (lane == 0) sc_s[n * 16 + b] = s;
    }
    __syncthreads();
    // ---- Phase C: softmax over n per b, then * mask ----
    if (tid < 16) {
      int b = tid;
      float mx = -3.0e38f;
      for (int n = 0; n < NN; ++n) mx = fmaxf(mx, sc_s[n * 16 + b]);
      float se = 0.f;
      for (int n = 0; n < NN; ++n) {
        float e = __expf(sc_s[n * 16 + b] - mx);
        sc_a[n * 16 + b] = e;
        se += e;
      }
      float inv = 1.f / se;
      for (int n = 0; n < NN; ++n)
        sc_a[n * 16 + b] *= inv * mask[((b * TT) + t) * NN + n];
    }
    __syncthreads();
    // ---- Phase D: attention[b][h] = sum_n alpha * obj ----
    for (int i = tid; i < 4096; i += 512) {
      int b = i >> 8, h = i & 255;
      float s = 0.f;
      for (int n = 0; n < NN; ++n)
        s += sc_a[n * 16 + b] *
             bfu2f(OBJ[(size_t)(((b * TT) + t) * NN + n) * HH + h]);
      qatt[b * 256 + h] = s;
    }
    // ---- Phase D2: risk outputs ----
    for (int p = wave; p < 640; p += 16) {
      int n = p >> 4, b = p & 15;
      const unsigned short* orow = OBJ + (size_t)(((b * TT) + t) * NN + n) * HH;
      float s = 0.f;
      for (int d = lane; d < HH; d += 32) s += bfu2f(orow[d]) * riskw[d];
      for (int off = 16; off; off >>= 1) s += __shfl_down(s, off, 32);
      if (lane == 0) {
        float sr = sc_a[n * 16 + b] * s + riskb[0];
        out_scr[((t * BB) + b) * NN + n] = sr;
        out_sigr[((b * TT) + t) * NN + n] = 1.f / (1.f + __expf(-sr));
      }
    }
    __syncthreads();
    // ---- Phase E+F: gates + LSTM update.  Wave w owns column tiles
    //      {w, w+16, w+32, w+48} => acc[j] is gate type j at cols [w*16, w*16+16)
    {
      v8f acc[4];
#pragma unroll
      for (int j = 0; j < 4; ++j)
#pragma unroll
        for (int v = 0; v < 8; ++v) acc[j][v] = 0.f;
      for (int k0 = 0; k0 < HH; k0 += 32) {
        v16bf aA = frag_from_lds_f32(qatt, lane, k0);  // attention
        v16bf aH = frag_from_lds_f32(hx, lane, k0);
#pragma unroll
        for (int j = 0; j < 4; ++j) {
          int ntile = j * 16 + wave;
          v16bf bI = load_frag(WIH + (size_t)ntile * 16 * HH, HH, lane, k0);
          acc[j] = WMMA_BF16(aA, bI, acc[j]);
          v16bf bH = load_frag(WHH + (size_t)ntile * 16 * HH, HH, lane, k0);
          acc[j] = WMMA_BF16(aH, bH, acc[j]);
        }
      }
      __syncthreads();   // everyone done reading hx before we overwrite it
      int roff = (lane >> 4) << 3;
      int h = wave * 16 + (lane & 15);
#pragma unroll
      for (int v = 0; v < 8; ++v) {
        int b = v + roff;
        float gi = acc[0][v] + bih[h] + bhh[h];
        float gf = acc[1][v] + bih[256 + h] + bhh[256 + h];
        float gg = acc[2][v] + bih[512 + h] + bhh[512 + h];
        float go = acc[3][v] + bih[768 + h] + bhh[768 + h];
        float c = sigf(gf) * cx[b * 256 + h] + sigf(gi) * tanhf(gg);
        cx[b * 256 + h] = c;
        hx[b * 256 + h] = sigf(go) * tanhf(c);
      }
    }
    __syncthreads();
    // ---- Phase G: collision head + softmax(2) ----
    if (tid < 32) {
      int b = tid >> 1, k = tid & 1;
      float s = collb[k];
      for (int h = 0; h < HH; ++h) s += hx[b * 256 + h] * collw[k * 256 + h];
      out_logc[(t * BB + b) * 2 + k] = s;
      tmpc[b * 2 + k] = s;
    }
    __syncthreads();
    if (tid < 16) {
      int b = tid;
      float s0 = tmpc[b * 2], s1 = tmpc[b * 2 + 1];
      float m = fmaxf(s0, s1);
      float e0 = __expf(s0 - m), e1 = __expf(s1 - m);
      float inv = 1.f / (e0 + e1);
      out_soft[(b * TT + t) * 2 + 0] = e0 * inv;
      out_soft[(b * TT + t) * 2 + 1] = e1 * inv;
    }
    __syncthreads();
  }
}

// ---------------------------------------------------------------------------
extern "C" void kernel_launch(void* const* d_in, const int* in_sizes, int n_in,
                              void* d_out, int out_size, void* d_ws, size_t ws_size,
                              hipStream_t stream) {
  (void)in_sizes; (void)n_in; (void)out_size; (void)ws_size;
  const float* in_feat = (const float*)d_in[0];
  const float* bn1_g = (const float*)d_in[2];
  const float* bn1_b = (const float*)d_in[3];
  const float* bn1_m = (const float*)d_in[4];
  const float* bn1_v = (const float*)d_in[5];
  const float* obj_w1 = (const float*)d_in[14];
  const float* obj_b1 = (const float*)d_in[15];
  const float* obn_g = (const float*)d_in[16];
  const float* obn_b = (const float*)d_in[17];
  const float* obn_m = (const float*)d_in[18];
  const float* obn_v = (const float*)d_in[19];
  const float* obj_w2 = (const float*)d_in[20];
  const float* obj_b2 = (const float*)d_in[21];
  const float* lstm_wih = (const float*)d_in[22];
  const float* lstm_whh = (const float*)d_in[23];
  const float* lstm_bih = (const float*)d_in[24];
  const float* lstm_bhh = (const float*)d_in[25];
  const float* coll_w = (const float*)d_in[26];
  const float* coll_b = (const float*)d_in[27];
  const float* risk_w = (const float*)d_in[28];
  const float* risk_b = (const float*)d_in[29];
  const float* att_w = (const float*)d_in[30];
  const float* att_wa = (const float*)d_in[31];
  const float* att_ua = (const float*)d_in[32];
  const float* att_ba = (const float*)d_in[33];

  char* ws = (char*)d_ws;
  unsigned short* X   = (unsigned short*)(ws + OFF_X);
  float*          IP  = (float*)(ws + OFF_X);        // alias: X dead after GEMM1
  unsigned short* H1  = (unsigned short*)(ws + OFF_H1);
  unsigned short* OBJ = (unsigned short*)(ws + OFF_OBJ);
  float*          MSK = (float*)(ws + OFF_MASK);
  unsigned short* W1  = (unsigned short*)(ws + OFF_W1);
  unsigned short* W2  = (unsigned short*)(ws + OFF_W2);
  unsigned short* WIH = (unsigned short*)(ws + OFF_WIH);
  unsigned short* WHH = (unsigned short*)(ws + OFF_WHH);
  unsigned short* UAT = (unsigned short*)(ws + OFF_UAT);
  unsigned short* WAT = (unsigned short*)(ws + OFF_WAT);
  float* SC1 = (float*)(ws + OFF_SC1);
  float* SH1 = (float*)(ws + OFF_SH1);

  k_prep_x<<<RR, 256, 0, stream>>>(in_feat, bn1_g, bn1_b, bn1_m, bn1_v, X, MSK);
  k_prep_w<<<4096, 256, 0, stream>>>(obj_w1, obj_w2, lstm_wih, lstm_whh, att_ua,
                                     att_wa, obj_b1, obn_g, obn_b, obn_m, obn_v,
                                     W1, W2, WIH, WHH, UAT, WAT, SC1, SH1);
  k_gemm1<<<dim3(300, 16), 256, 0, stream>>>(X, W1, SC1, SH1, H1);
  k_gemm2<<<dim3(300, 4), 256, 0, stream>>>(H1, W2, obj_b2, MSK, OBJ);
  k_gemm_ip<<<dim3(300, 4), 256, 0, stream>>>(OBJ, UAT, att_ba, IP);
  k_scan<<<1, 512, 0, stream>>>(OBJ, IP, WAT, WIH, WHH, att_w, MSK, lstm_bih,
                                lstm_bhh, coll_w, coll_b, risk_w, risk_b,
                                (float*)d_out);
}
